// OLMo3Attention_13469017440720
// MI455X (gfx1250) — matrix operand order repaired and checked by reading, not compiled, not run on hardware
//
#include <hip/hip_runtime.h>

// ---------------------------------------------------------------------------
// OLMo3 sliding-window attention for MI455X (gfx1250, wave32, WMMA).
// All three matmul stages (QKV proj, QK^T/PV, O proj) run on
// v_wmma_f32_16x16x32_bf16 (fp32 operands pre-converted to bf16, fp32 acc).
// Attention is flash-style over the 1024-wide band only (65 of 128 k-tiles).
// ---------------------------------------------------------------------------

#define S_LEN  2048
#define DMODEL 2048
#define NH     16
#define NKV    4
#define HD     128
#define KVDIM  (NKV * HD)           // 512
#define NQKV   (DMODEL + 2 * KVDIM) // 3072
#define MROWS  (2 * S_LEN)          // 4096 (batch-flattened)

typedef __attribute__((ext_vector_type(16))) __bf16 v16bf;
typedef __attribute__((ext_vector_type(8)))  __bf16 v8bf;
typedef __attribute__((ext_vector_type(8)))  float  v8f;

static __device__ __forceinline__ unsigned short f2bf(float f) {
  unsigned u = __float_as_uint(f);
  unsigned r = u + 0x7FFFu + ((u >> 16) & 1u);   // round to nearest even
  return (unsigned short)(r >> 16);
}

static __device__ __forceinline__ v8f wmma_bf16(v16bf a, v16bf b, v8f c) {
  return __builtin_amdgcn_wmma_f32_16x16x32_bf16(false, a, false, b,
                                                 (short)0, c, false, false);
}

// A fragment (16x32, row-major source): lane L holds row m0+(L&15);
// two contiguous 8-elem chunks at k0+16*(L>>4) and k0+16*(L>>4)+8.
static __device__ __forceinline__ v16bf load_afrag(const unsigned short* A,
                                                   int ld, int m0, int k0,
                                                   int lane) {
  int m  = m0 + (lane & 15);
  int kb = k0 + 16 * (lane >> 4);
  const unsigned short* p = A + (size_t)m * ld + kb;
  v8bf lo = *(const v8bf*)(p);
  v8bf hi = *(const v8bf*)(p + 8);
  return __builtin_shufflevector(lo, hi, 0, 1, 2, 3, 4, 5, 6, 7,
                                 8, 9, 10, 11, 12, 13, 14, 15);
}

// B fragment (32x16): lane L holds column n0+(L&15) of B = row of W[N,K];
// one contiguous 16-elem chunk at k0+16*(L>>4).  (32B aligned load)
static __device__ __forceinline__ v16bf load_bfrag(const unsigned short* W,
                                                   int ld, int n0, int k0,
                                                   int lane) {
  int n  = n0 + (lane & 15);
  int kb = k0 + 16 * (lane >> 4);
  return *(const v16bf*)(W + (size_t)n * ld + kb);
}

static __device__ __forceinline__ v16bf load_bfrag_clampn(const unsigned short* W,
                                                          int ld, int n0, int k0,
                                                          int nmax, int lane) {
  int n = n0 + (lane & 15);
  n = (n > nmax) ? nmax : n;
  int kb = k0 + 16 * (lane >> 4);
  return *(const v16bf*)(W + (size_t)n * ld + kb);
}

// V fragment from V^T [hd, S]: rows = dims, K-dim = keys (clamped to stay in range)
static __device__ __forceinline__ v16bf load_vfrag(const unsigned short* Vh,
                                                   int d0, int kabs, int lane) {
  int dim = d0 + (lane & 15);
  int kb  = kabs + 16 * (lane >> 4);
  if (kb > S_LEN - 16) kb = S_LEN - 16;
  return *(const v16bf*)(Vh + (size_t)dim * S_LEN + kb);
}

// ---------------------------------------------------------------------------
__global__ __launch_bounds__(256) void cvt_bf16(const float* __restrict__ src,
                                                unsigned short* __restrict__ dst,
                                                int n) {
  int i = blockIdx.x * blockDim.x + threadIdx.x;
  if (i < n) dst[i] = f2bf(src[i]);
}

// ---------------------------------------------------------------------------
// C[M,N] = A[M,K] @ W[N,K]^T   (bf16 in, fp32 out)
// One wave computes a 32(M) x 64(N) tile; block = 4 waves along N.
__global__ __launch_bounds__(128) void gemm_bf16(const unsigned short* __restrict__ A,
                                                 const unsigned short* __restrict__ W,
                                                 float* __restrict__ C,
                                                 int N, int K) {
  int lane = threadIdx.x & 31;
  int wid  = threadIdx.x >> 5;
  int m0   = blockIdx.x * 32;
  int n0   = blockIdx.y * 256 + wid * 64;

  v8f acc[8] = {};
  for (int k0 = 0; k0 < K; k0 += 32) {
    v16bf a0 = load_afrag(A, K, m0,      k0, lane);
    v16bf a1 = load_afrag(A, K, m0 + 16, k0, lane);
#pragma unroll
    for (int nt = 0; nt < 4; ++nt) {
      v16bf bf = load_bfrag(W, K, n0 + nt * 16, k0, lane);
      acc[nt]     = wmma_bf16(a0, bf, acc[nt]);
      acc[nt + 4] = wmma_bf16(a1, bf, acc[nt + 4]);
    }
  }
  int ml = 8 * (lane >> 4);
  int nl = lane & 15;
#pragma unroll
  for (int nt = 0; nt < 4; ++nt)
#pragma unroll
    for (int r = 0; r < 8; ++r) {
      C[(size_t)(m0 + r + ml) * N + n0 + nt * 16 + nl]      = acc[nt][r];
      C[(size_t)(m0 + 16 + r + ml) * N + n0 + nt * 16 + nl] = acc[nt + 4][r];
    }
}

// ---------------------------------------------------------------------------
// RoPE epilogue: QKVf fp32 [MROWS, 3072] -> Qb [B,H,S,hd] (pre-scaled by
// 1/sqrt(hd)), Kb [B,KV,S,hd], both bf16.
__global__ __launch_bounds__(256) void rope_qk(const float* __restrict__ qkv,
                                               const int* __restrict__ pos,
                                               unsigned short* __restrict__ Qb,
                                               unsigned short* __restrict__ Kb) {
  const int NQ = 2 * S_LEN * NH * 64;
  const int NK = 2 * S_LEN * NKV * 64;
  int idx = blockIdx.x * blockDim.x + threadIdx.x;
  if (idx >= NQ + NK) return;
  const float LOG_THETA = 13.122363377403645f;  // ln(500000)
  if (idx < NQ) {
    int d = idx & 63; int t = idx >> 6;
    int h = t & 15;   t >>= 4;
    int s = t & (S_LEN - 1);
    int b = t >> 11;
    const float* row = qkv + (size_t)(b * S_LEN + s) * NQKV + h * HD;
    float x1 = row[d], x2 = row[d + 64];
    float ang = (float)pos[s] * __expf(-(float)d * (LOG_THETA / 64.0f));
    float sn, cs; __sincosf(ang, &sn, &cs);
    const float qs = 0.088388347648318447f;     // 1/sqrt(128), folded into Q
    unsigned short* o = Qb + (size_t)((b * NH + h) * S_LEN + s) * HD;
    o[d]      = f2bf((x1 * cs - x2 * sn) * qs);
    o[d + 64] = f2bf((x2 * cs + x1 * sn) * qs);
  } else {
    idx -= NQ;
    int d = idx & 63; int t = idx >> 6;
    int kvh = t & 3;  t >>= 2;
    int s = t & (S_LEN - 1);
    int b = t >> 11;
    const float* row = qkv + (size_t)(b * S_LEN + s) * NQKV + DMODEL + kvh * HD;
    float x1 = row[d], x2 = row[d + 64];
    float ang = (float)pos[s] * __expf(-(float)d * (LOG_THETA / 64.0f));
    float sn, cs; __sincosf(ang, &sn, &cs);
    unsigned short* o = Kb + (size_t)((b * NKV + kvh) * S_LEN + s) * HD;
    o[d]      = f2bf(x1 * cs - x2 * sn);
    o[d + 64] = f2bf(x2 * cs + x1 * sn);
  }
}

// V -> bf16, transposed to [B,KV,hd,S] so PV B-fragments are contiguous.
__global__ __launch_bounds__(256) void v_transpose(const float* __restrict__ qkv,
                                                   unsigned short* __restrict__ Vt) {
  int idx = blockIdx.x * blockDim.x + threadIdx.x;
  if (idx >= 2 * NKV * HD * S_LEN) return;
  int s = idx & (S_LEN - 1); int t = idx >> 11;
  int d = t & 127;           t >>= 7;
  int kvh = t & 3;
  int b = t >> 2;
  float v = qkv[(size_t)(b * S_LEN + s) * NQKV + DMODEL + KVDIM + kvh * HD + d];
  Vt[(size_t)((b * NKV + kvh) * HD + d) * S_LEN + s] = f2bf(v);
}

// ---------------------------------------------------------------------------
// Flash attention over the sliding band: one wave per (b, h, 16-query tile).
// Mask: allowed iff 0 <= key - query < window (and key < S).  The band for a
// query tile spans 65 key tiles -> 33 steps of 32 keys each.
__global__ __launch_bounds__(128) void attn_fa(const unsigned short* __restrict__ Qb,
                                               const unsigned short* __restrict__ Kb,
                                               const unsigned short* __restrict__ Vt,
                                               unsigned short* __restrict__ attnb,
                                               const int* __restrict__ winp) {
  __shared__ unsigned short plds[4][16 * 40];   // P tile per wave, stride 40 (80B rows)

  int lane = threadIdx.x & 31;
  int wid  = threadIdx.x >> 5;
  int gw   = blockIdx.x * 4 + wid;
  int qt = gw & 127;
  int h  = (gw >> 7) & 15;
  int b  = gw >> 11;
  int q0 = qt * 16;
  int kvh = h >> 2;                              // repeat_interleave(groups=4)
  int window = winp[0];

  const unsigned short* Qh = Qb + (size_t)(b * NH + h) * S_LEN * HD;
  const unsigned short* Kh = Kb + (size_t)(b * NKV + kvh) * S_LEN * HD;
  const unsigned short* Vh = Vt + (size_t)(b * NKV + kvh) * HD * S_LEN;

  v16bf qf[4];
#pragma unroll
  for (int kk = 0; kk < 4; ++kk) qf[kk] = load_afrag(Qh, HD, q0, kk * 32, lane);

  v8f acc[8] = {};
  float rmax[8], rsum[8];
#pragma unroll
  for (int r = 0; r < 8; ++r) { rmax[r] = -1e30f; rsum[r] = 0.0f; }

  int mhalf = 8 * (lane >> 4);
  int nl    = lane & 15;

  for (int step = 0; step < 33; ++step) {
    int k0 = q0 + step * 32;
    if (k0 >= S_LEN) break;

    // scores: two 16x16 tiles (keys k0.., k0+16..), K-dim = hd = 4 WMMA each
    v8f s0 = {}, s1 = {};
#pragma unroll
    for (int kk = 0; kk < 4; ++kk) {
      v16bf kf0 = load_bfrag_clampn(Kh, HD, k0,      kk * 32, S_LEN - 1, lane);
      s0 = wmma_bf16(qf[kk], kf0, s0);
      v16bf kf1 = load_bfrag_clampn(Kh, HD, k0 + 16, kk * 32, S_LEN - 1, lane);
      s1 = wmma_bf16(qf[kk], kf1, s1);
    }

    int keyA = k0 + nl;
    int keyB = keyA + 16;
#pragma unroll
    for (int r = 0; r < 8; ++r) {
      int q = q0 + r + mhalf;
      int dA = keyA - q, dB = keyB - q;
      if (!(dA >= 0 && dA < window && keyA < S_LEN)) s0[r] = -1e30f;
      if (!(dB >= 0 && dB < window && keyB < S_LEN)) s1[r] = -1e30f;
    }

    // online softmax per row (rows live in 16-lane groups -> xor-shuffle <16)
#pragma unroll
    for (int r = 0; r < 8; ++r) {
      float tm = fmaxf(s0[r], s1[r]);
#pragma unroll
      for (int off = 1; off < 16; off <<= 1) tm = fmaxf(tm, __shfl_xor(tm, off, 32));
      float nm = fmaxf(rmax[r], tm);
      float p0 = __expf(s0[r] - nm);
      float p1 = __expf(s1[r] - nm);
      float ts = p0 + p1;
#pragma unroll
      for (int off = 1; off < 16; off <<= 1) ts += __shfl_xor(ts, off, 32);
      float sc = __expf(rmax[r] - nm);
      rsum[r] = rsum[r] * sc + ts;
      rmax[r] = nm;
#pragma unroll
      for (int d = 0; d < 8; ++d) acc[d][r] *= sc;
      int row = r + mhalf;
      plds[wid][row * 40 + nl]      = f2bf(p0);   // D-layout -> row-major in LDS
      plds[wid][row * 40 + 16 + nl] = f2bf(p1);
    }

    // read P back in A-layout (LDS is in-order within a wave)
    {
      const unsigned short* prow = &plds[wid][(lane & 15) * 40];
      int c0 = 16 * (lane >> 4);
      v8bf lo = *(const v8bf*)(prow + c0);
      v8bf hi = *(const v8bf*)(prow + c0 + 8);
      v16bf pf = __builtin_shufflevector(lo, hi, 0, 1, 2, 3, 4, 5, 6, 7,
                                         8, 9, 10, 11, 12, 13, 14, 15);
#pragma unroll
      for (int dt = 0; dt < 8; ++dt) {
        v16bf vf = load_vfrag(Vh, dt * 16, k0, lane);
        acc[dt] = wmma_bf16(pf, vf, acc[dt]);
      }
    }
  }

  // normalize + store bf16 to attn [B, S, H*hd]
#pragma unroll
  for (int r = 0; r < 8; ++r) {
    float inv = 1.0f / rsum[r];
    int q = q0 + r + mhalf;
    unsigned short* orow = attnb + (size_t)(b * S_LEN + q) * DMODEL + h * HD;
#pragma unroll
    for (int dt = 0; dt < 8; ++dt)
      orow[dt * 16 + nl] = f2bf(acc[dt][r] * inv);
  }
}

// ---------------------------------------------------------------------------
extern "C" void kernel_launch(void* const* d_in, const int* in_sizes, int n_in,
                              void* d_out, int out_size, void* d_ws, size_t ws_size,
                              hipStream_t stream) {
  const float* x   = (const float*)d_in[0];   // [2,2048,2048]
  const float* q_w = (const float*)d_in[1];   // [2048,2048]
  const float* k_w = (const float*)d_in[2];   // [512,2048]
  const float* v_w = (const float*)d_in[3];   // [512,2048]
  const float* o_w = (const float*)d_in[4];   // [2048,2048]
  const int*   pos = (const int*)d_in[5];     // [2048]
  const int*   win = (const int*)d_in[6];     // scalar
  float* out = (float*)d_out;

  char* ws = (char*)d_ws;
  unsigned short* Xb    = (unsigned short*)(ws);                     // 16.0 MiB
  unsigned short* Wqkvb = (unsigned short*)(ws + 16777216);          // 12.0 MiB
  unsigned short* o_wb  = (unsigned short*)(ws + 29360128);          //  8.0 MiB
  unsigned short* Qb    = (unsigned short*)(ws + 37748736);          // 16.0 MiB
  unsigned short* Kb    = (unsigned short*)(ws + 54525952);          //  4.0 MiB
  unsigned short* Vt    = (unsigned short*)(ws + 58720256);          //  4.0 MiB
  unsigned short* attnb = (unsigned short*)(ws + 62914560);          // 16 MiB (aliases QKVf head)
  float*          QKVf  = (float*)(ws + 62914560);                   // 48 MiB (dead before attnb written)

  // fp32 -> bf16 conversions (weights packed as [q;k;v] rows)
  {
    int n;
    n = MROWS * DMODEL;   cvt_bf16<<<(n + 255) / 256, 256, 0, stream>>>(x,   Xb, n);
    n = DMODEL * DMODEL;  cvt_bf16<<<(n + 255) / 256, 256, 0, stream>>>(q_w, Wqkvb, n);
    n = KVDIM * DMODEL;   cvt_bf16<<<(n + 255) / 256, 256, 0, stream>>>(k_w, Wqkvb + (size_t)DMODEL * DMODEL, n);
    n = KVDIM * DMODEL;   cvt_bf16<<<(n + 255) / 256, 256, 0, stream>>>(v_w, Wqkvb + (size_t)(DMODEL + KVDIM) * DMODEL, n);
    n = DMODEL * DMODEL;  cvt_bf16<<<(n + 255) / 256, 256, 0, stream>>>(o_w, o_wb, n);
  }

  // fused QKV projection: [4096,2048] @ [3072,2048]^T -> fp32 [4096,3072]
  gemm_bf16<<<dim3(MROWS / 32, NQKV / 256), 128, 0, stream>>>(Xb, Wqkvb, QKVf, NQKV, DMODEL);

  // RoPE + layout epilogues
  {
    int n = 2 * S_LEN * NH * 64 + 2 * S_LEN * NKV * 64;
    rope_qk<<<(n + 255) / 256, 256, 0, stream>>>(QKVf, pos, Qb, Kb);
    n = 2 * NKV * HD * S_LEN;
    v_transpose<<<(n + 255) / 256, 256, 0, stream>>>(QKVf, Vt);
  }

  // banded flash attention: 2*16*128 = 4096 waves, 4 per block
  attn_fa<<<1024, 128, 0, stream>>>(Qb, Kb, Vt, attnb, win);

  // output projection: [4096,2048] @ [2048,2048]^T -> fp32 d_out
  gemm_bf16<<<dim3(MROWS / 32, DMODEL / 256), 128, 0, stream>>>(attnb, o_wb, out, DMODEL, DMODEL);
}
